// TransformerBlock_17669495456357
// MI455X (gfx1250) — compile-verified, matrix operand order checked by reading, not run on hardware
//
#include <hip/hip_runtime.h>
#include <hip/hip_bf16.h>

// ---------------- constants (match reference) ----------------
#define GG 3
#define BB 4
#define TT 512
#define NN 512
#define DD 1024
#define EE 1024
#define FF 4096
#define HH 16
#define HDD 64
#define GM 2048            // rows per branch in every big GEMM (B*T = B*N)
#define KVTOT (GG*NN)      // 1536
#define QTOT  (GG*TT)      // 1536

typedef __attribute__((ext_vector_type(16))) __bf16 v16bf;
typedef __attribute__((ext_vector_type(8)))  float  v8f;

union FragU { v16bf v; uint4 q[2]; };
union U8    { uint4 q; __bf16 e[8]; };

__device__ __forceinline__ v8f wmma_bf16(v16bf a, v16bf b, v8f c) {
    return __builtin_amdgcn_wmma_f32_16x16x32_bf16(
        /*neg_a=*/false, a, /*neg_b=*/false, b,
        /*c_mod=*/(short)0, c, /*reuse_a=*/false, /*reuse_b=*/false);
}

// ---- gfx1250 async global->LDS copy (bypasses VGPRs, tracked by ASYNCcnt) ----
// LDS VDST operand wants the wave-relative LDS byte address: generic->AS(3)
// addrspacecast truncates to the 32-bit LDS offset.
__device__ __forceinline__ unsigned lds_off(const void* p) {
    return (unsigned)(unsigned long long)(const __attribute__((address_space(3))) char*)p;
}
__device__ __forceinline__ void async_ld128(const void* gsrc, const void* ldst) {
    unsigned l = lds_off(ldst);
    asm volatile("global_load_async_to_lds_b128 %0, %1, off"
                 :: "v"(l), "v"(gsrc) : "memory");
}
__device__ __forceinline__ void wait_async0() {
    asm volatile("s_wait_asynccnt 0" ::: "memory");
}

// ---------------- fp32 -> bf16 conversion ----------------
__global__ void cvt_bf16_kernel(const float* __restrict__ in, __bf16* __restrict__ out, int n) {
    int i = (blockIdx.x * 256 + threadIdx.x) * 4;
    if (i + 3 < n) {
        float4 v = *(const float4*)(in + i);
        out[i + 0] = (__bf16)v.x; out[i + 1] = (__bf16)v.y;
        out[i + 2] = (__bf16)v.z; out[i + 3] = (__bf16)v.w;
    }
}

// ---------------- silu(emb) ----------------
__global__ void silu_kernel(const float* __restrict__ emb, float* __restrict__ out, int n) {
    int i = blockIdx.x * 256 + threadIdx.x;
    if (i < n) {
        float v = emb[i];
        out[i] = v / (1.0f + __expf(-v));
    }
}

// ---------------- modulation: out[g,b,c] = silu(emb)[b,:] . W[g,c,:] + bias[g,c] ----------------
__global__ void mod_kernel(const float* __restrict__ sil, const float* __restrict__ W,
                           const float* __restrict__ bias, float* __restrict__ out,
                           int C2) {
    int idx = blockIdx.x * 256 + threadIdx.x;           // over G*B*C2
    int c  = idx % C2;
    int gb = idx / C2;
    int b  = gb % BB;
    int g  = gb / BB;
    const float* s = sil + (size_t)b * EE;
    const float* w = W + ((size_t)g * C2 + c) * EE;
    float acc = 0.f;
    #pragma unroll 4
    for (int e = 0; e < EE; e += 4) {
        float4 sv = *(const float4*)(s + e);
        float4 wv = *(const float4*)(w + e);
        acc += sv.x * wv.x + sv.y * wv.y + sv.z * wv.z + sv.w * wv.w;
    }
    out[idx] = acc + bias[(size_t)g * C2 + c];
}

// ---------------- fused LayerNorm + AdaLN -> bf16 ----------------
// h: [G,B,512,1024] fp32 row = blockIdx.x over G*B*512 ; mod: [G,B,2048]
__global__ void adaln_kernel(const float* __restrict__ h, const float* __restrict__ mod,
                             __bf16* __restrict__ out) {
    __shared__ float red[256];
    int row = blockIdx.x;
    int tid = threadIdx.x;
    int gb  = row / 512;             // (g*B + b)
    size_t base = (size_t)row * DD;
    float4 xv = *(const float4*)(h + base + tid * 4);
    float s  = xv.x + xv.y + xv.z + xv.w;
    float s2 = xv.x * xv.x + xv.y * xv.y + xv.z * xv.z + xv.w * xv.w;

    red[tid] = s; __syncthreads();
    for (int st = 128; st > 0; st >>= 1) { if (tid < st) red[tid] += red[tid + st]; __syncthreads(); }
    float mean = red[0] * (1.0f / DD);
    __syncthreads();
    red[tid] = s2; __syncthreads();
    for (int st = 128; st > 0; st >>= 1) { if (tid < st) red[tid] += red[tid + st]; __syncthreads(); }
    float var = red[0] * (1.0f / DD) - mean * mean;
    float rs  = rsqrtf(var + 1e-6f);

    const float* m = mod + (size_t)gb * (2 * DD);
    int c = tid * 4;
    float vals[4] = { xv.x, xv.y, xv.z, xv.w };
    #pragma unroll
    for (int j = 0; j < 4; ++j) {
        float sc = m[c + j];
        float sh = m[DD + c + j];
        float v  = (vals[j] - mean) * rs * (1.0f + sc) + sh;
        out[base + c + j] = (__bf16)v;
    }
}

// ---------------- tiled WMMA GEMM with async global->LDS double buffering ----------------
// per g (blockIdx.z): out[m,n] = act( (A[m,:] . W[n,:] + bias[n]) * scale ) + resid[m,n]
// A: bf16 [GM x K] (g-major), W: bf16 [N x K] (g-major), bias fp32 [N]
__global__ __launch_bounds__(256)
void gemm_bf16_kernel(const __bf16* __restrict__ A, const __bf16* __restrict__ W,
                      const float* __restrict__ bias, const float* __restrict__ resid,
                      float scale, int act,
                      __bf16* __restrict__ outb, float* __restrict__ outf,
                      int Nn, int Kk) {
    __shared__ __bf16 As[2][128 * 40];
    __shared__ __bf16 Bs[2][128 * 40];

    int g  = blockIdx.z;
    int m0 = blockIdx.y * 128;
    int n0 = blockIdx.x * 128;
    int tid = threadIdx.x;
    int wave = tid >> 5, lane = tid & 31;
    int laneM = lane & 15, kh = lane >> 4;
    int wm = wave >> 2, wn = wave & 3;      // 2 x 4 wave grid -> 64 x 32 per wave

    const __bf16* Ag = A + (size_t)g * GM * Kk + (size_t)m0 * Kk;
    const __bf16* Wg = W + (size_t)g * Nn * Kk + (size_t)n0 * Kk;
    const float*  bg = bias + (size_t)g * Nn;
    const float*  rg = resid ? resid + ((size_t)g * GM + m0) * Nn : nullptr;

    int s0 = tid, s1 = tid + 256;
    int ar0 = s0 >> 2, ac0 = (s0 & 3) * 8;
    int ar1 = s1 >> 2, ac1 = (s1 & 3) * 8;

    int KT = Kk >> 5;  // K / 32

    // async copy of one 128x32 A tile + 128x32 W tile into buffer `buf`
    auto tile_async = [&](int kt, int buf) {
        int k0 = kt * 32;
        async_ld128(Ag + (size_t)ar0 * Kk + k0 + ac0, &As[buf][ar0 * 40 + ac0]);
        async_ld128(Ag + (size_t)ar1 * Kk + k0 + ac1, &As[buf][ar1 * 40 + ac1]);
        async_ld128(Wg + (size_t)ar0 * Kk + k0 + ac0, &Bs[buf][ar0 * 40 + ac0]);
        async_ld128(Wg + (size_t)ar1 * Kk + k0 + ac1, &Bs[buf][ar1 * 40 + ac1]);
    };

    // prologue: tile 0 -> buffer 0
    tile_async(0, 0);
    wait_async0();
    __syncthreads();

    v8f acc[4][2];
    #pragma unroll
    for (int mt = 0; mt < 4; ++mt)
        #pragma unroll
        for (int nt = 0; nt < 2; ++nt)
            acc[mt][nt] = (v8f){0.f,0.f,0.f,0.f,0.f,0.f,0.f,0.f};

    for (int kt = 0; kt < KT; ++kt) {
        int cur = kt & 1;
        if (kt + 1 < KT) tile_async(kt + 1, cur ^ 1);   // async into other buffer
        if (kt + 2 < KT) {
            int k2 = (kt + 2) * 32;
            __builtin_prefetch(Ag + (size_t)ar0 * Kk + k2 + ac0, 0, 1);
            __builtin_prefetch(Wg + (size_t)ar0 * Kk + k2 + ac0, 0, 1);
        }

        // fragments
        FragU aF[4], bF[2];
        #pragma unroll
        for (int mt = 0; mt < 4; ++mt) {
            const __bf16* p = &As[cur][(wm * 64 + mt * 16 + laneM) * 40];
            aF[mt].q[0] = *(const uint4*)(p + kh * 8);
            aF[mt].q[1] = *(const uint4*)(p + 16 + kh * 8);
        }
        #pragma unroll
        for (int nt = 0; nt < 2; ++nt) {
            const __bf16* p = &Bs[cur][(wn * 32 + nt * 16 + laneM) * 40];
            bF[nt].q[0] = *(const uint4*)(p + kh * 8);
            bF[nt].q[1] = *(const uint4*)(p + 16 + kh * 8);
        }
        #pragma unroll
        for (int mt = 0; mt < 4; ++mt)
            #pragma unroll
            for (int nt = 0; nt < 2; ++nt)
                acc[mt][nt] = wmma_bf16(aF[mt].v, bF[nt].v, acc[mt][nt]);

        if (kt + 1 < KT) wait_async0();   // next buffer resident before barrier
        __syncthreads();
    }

    // epilogue
    #pragma unroll
    for (int mt = 0; mt < 4; ++mt) {
        #pragma unroll
        for (int nt = 0; nt < 2; ++nt) {
            int ocol = n0 + wn * 32 + nt * 16 + laneM;
            float bval = bg[ocol];
            #pragma unroll
            for (int r = 0; r < 8; ++r) {
                int orow = m0 + wm * 64 + mt * 16 + r + 8 * kh;
                float v = (acc[mt][nt][r] + bval) * scale;
                if (act) v = 0.5f * v * (1.0f + erff(v * 0.70710678118f));
                if (rg)  v += rg[(size_t)(orow - m0) * Nn + ocol];
                size_t o = ((size_t)g * GM + orow) * Nn + ocol;
                if (outf) outf[o] = v;
                if (outb) outb[o] = (__bf16)v;
            }
        }
    }
}

// ---------------- flash attention (bf16 WMMA, online softmax) ----------------
// Q/K/V/O: bf16 [G,B,512,1024]; logical Qh[b,h,q=g*T+t, hd] = Q[g,b,t,h*64+hd]
// grid: (QTOT/128, H, B), 256 threads = 8 waves, 16 q-rows per wave, KV chunk = 64
__global__ __launch_bounds__(256)
void attn_kernel(const __bf16* __restrict__ Qb, const __bf16* __restrict__ Kb,
                 const __bf16* __restrict__ Vb, __bf16* __restrict__ Ob) {
    __shared__ __bf16 Ks[64 * 72];         // [kv][hd]
    __shared__ __bf16 Vts[64 * 72];        // [hd][kv] (transposed)
    __shared__ __bf16 Ps[8 * 16 * 72];     // per-wave P staging [wave][16][72]

    int tid = threadIdx.x;
    int wave = tid >> 5, lane = tid & 31;
    int laneM = lane & 15, kh = lane >> 4;
    int h = blockIdx.y, b = blockIdx.z;

    // Q fragments (16 rows x 64 hd) for this wave, kept in registers
    int qrow = blockIdx.x * 128 + wave * 16 + laneM;
    int gq = qrow >> 9, tq = qrow & 511;
    const __bf16* qp = Qb + ((((size_t)gq * BB + b) * TT + tq) * DD + h * HDD);
    FragU qa0, qa1;
    qa0.q[0] = *(const uint4*)(qp + kh * 8);
    qa0.q[1] = *(const uint4*)(qp + 16 + kh * 8);
    qa1.q[0] = *(const uint4*)(qp + 32 + kh * 8);
    qa1.q[1] = *(const uint4*)(qp + 48 + kh * 8);

    float m_i[8], l_i[8];
    v8f accO[4];
    #pragma unroll
    for (int r = 0; r < 8; ++r) { m_i[r] = -1e30f; l_i[r] = 0.f; }
    #pragma unroll
    for (int t = 0; t < 4; ++t) accO[t] = (v8f){0.f,0.f,0.f,0.f,0.f,0.f,0.f,0.f};

    for (int c = 0; c < KVTOT / 64; ++c) {
        if (c) __syncthreads();
        // load K chunk ([kv][hd]) and V chunk transposed ([hd][kv])
        for (int s = tid; s < 512; s += 256) {
            int row = s >> 3, c8 = (s & 7) * 8;
            int kvg = c * 64 + row;
            int gk = kvg >> 9, nk = kvg & 511;
            size_t src = (((size_t)gk * BB + b) * NN + nk) * DD + h * HDD + c8;
            *(uint4*)(&Ks[row * 72 + c8]) = *(const uint4*)(Kb + src);
            U8 vv; vv.q = *(const uint4*)(Vb + src);
            #pragma unroll
            for (int j = 0; j < 8; ++j) Vts[(c8 + j) * 72 + row] = vv.e[j];
        }
        __syncthreads();

        // S = Q . K^T  (4 tiles of 16x16 over 64 kv)
        v8f S[4];
        #pragma unroll
        for (int nt = 0; nt < 4; ++nt) {
            const __bf16* kp = &Ks[(nt * 16 + laneM) * 72];
            FragU b0, b1;
            b0.q[0] = *(const uint4*)(kp + kh * 8);
            b0.q[1] = *(const uint4*)(kp + 16 + kh * 8);
            b1.q[0] = *(const uint4*)(kp + 32 + kh * 8);
            b1.q[1] = *(const uint4*)(kp + 48 + kh * 8);
            v8f z = (v8f){0.f,0.f,0.f,0.f,0.f,0.f,0.f,0.f};
            z = wmma_bf16(qa0.v, b0.v, z);
            S[nt] = wmma_bf16(qa1.v, b1.v, z);
        }

        // online softmax update (rows M = r + 8*kh, 16-lane butterflies)
        __bf16* pw = &Ps[wave * 16 * 72];
        #pragma unroll
        for (int r = 0; r < 8; ++r) {
            float mx = fmaxf(fmaxf(S[0][r], S[1][r]), fmaxf(S[2][r], S[3][r]));
            mx = fmaxf(mx, __shfl_xor(mx, 1, 32));
            mx = fmaxf(mx, __shfl_xor(mx, 2, 32));
            mx = fmaxf(mx, __shfl_xor(mx, 4, 32));
            mx = fmaxf(mx, __shfl_xor(mx, 8, 32));
            float mnew = fmaxf(m_i[r], mx);
            float sc = __expf(m_i[r] - mnew);
            m_i[r] = mnew;
            int M = r + 8 * kh;
            float ls = 0.f;
            #pragma unroll
            for (int nt = 0; nt < 4; ++nt) {
                float p = __expf(S[nt][r] - mnew);
                ls += p;
                pw[M * 72 + nt * 16 + laneM] = (__bf16)p;
            }
            ls += __shfl_xor(ls, 1, 32);
            ls += __shfl_xor(ls, 2, 32);
            ls += __shfl_xor(ls, 4, 32);
            ls += __shfl_xor(ls, 8, 32);
            l_i[r] = l_i[r] * sc + ls;
            #pragma unroll
            for (int t = 0; t < 4; ++t) accO[t][r] *= sc;
        }

        // O += P . V  (P: 16x64 via LDS restage; V^T tiles give contiguous kv chunks)
        const __bf16* pp = &Ps[wave * 16 * 72 + laneM * 72];
        FragU pa0, pa1;
        pa0.q[0] = *(const uint4*)(pp + kh * 8);
        pa0.q[1] = *(const uint4*)(pp + 16 + kh * 8);
        pa1.q[0] = *(const uint4*)(pp + 32 + kh * 8);
        pa1.q[1] = *(const uint4*)(pp + 48 + kh * 8);
        #pragma unroll
        for (int hdt = 0; hdt < 4; ++hdt) {
            const __bf16* vp = &Vts[(hdt * 16 + laneM) * 72];
            FragU v0, v1;
            v0.q[0] = *(const uint4*)(vp + kh * 8);
            v0.q[1] = *(const uint4*)(vp + 16 + kh * 8);
            v1.q[0] = *(const uint4*)(vp + 32 + kh * 8);
            v1.q[1] = *(const uint4*)(vp + 48 + kh * 8);
            accO[hdt] = wmma_bf16(pa0.v, v0.v, accO[hdt]);
            accO[hdt] = wmma_bf16(pa1.v, v1.v, accO[hdt]);
        }
    }

    // write O (C layout: M = r + 8*kh, col = hdt*16 + laneM)
    #pragma unroll
    for (int hdt = 0; hdt < 4; ++hdt) {
        int hd = hdt * 16 + laneM;
        #pragma unroll
        for (int r = 0; r < 8; ++r) {
            int M = r + 8 * kh;
            int q = blockIdx.x * 128 + wave * 16 + M;
            int g = q >> 9, t = q & 511;
            float v = accO[hdt][r] / l_i[r];
            Ob[(((size_t)g * BB + b) * TT + t) * DD + h * HDD + hd] = (__bf16)v;
        }
    }
}

// ---------------- host-side launcher ----------------
extern "C" void kernel_launch(void* const* d_in, const int* in_sizes, int n_in,
                              void* d_out, int out_size, void* d_ws, size_t ws_size,
                              hipStream_t stream) {
    const float* x       = (const float*)d_in[0];
    const float* xf      = (const float*)d_in[1];
    const float* emb     = (const float*)d_in[2];
    const float* Wemb_x  = (const float*)d_in[3];
    const float* bemb_x  = (const float*)d_in[4];
    const float* Wemb_xf = (const float*)d_in[5];
    const float* bemb_xf = (const float*)d_in[6];
    const float* Wq = (const float*)d_in[7];  const float* bq = (const float*)d_in[8];
    const float* Wk = (const float*)d_in[9];  const float* bk = (const float*)d_in[10];
    const float* Wv = (const float*)d_in[11]; const float* bv = (const float*)d_in[12];
    const float* Wo = (const float*)d_in[13]; const float* bo = (const float*)d_in[14];
    const float* Wemb_f = (const float*)d_in[15]; const float* bemb_f = (const float*)d_in[16];
    const float* W1 = (const float*)d_in[17]; const float* b1 = (const float*)d_in[18];
    const float* W2 = (const float*)d_in[19]; const float* b2 = (const float*)d_in[20];
    (void)in_sizes; (void)n_in; (void)out_size; (void)ws_size;

    char* ws = (char*)d_ws;
    size_t off = 0;
    auto take = [&](size_t bytes) -> char* {
        char* p = ws + off;
        off += (bytes + 255) & ~(size_t)255;
        return p;
    };

    const size_t WSQ = (size_t)GG * DD * DD;       // 3*1024*1024
    const size_t WF  = (size_t)GG * FF * DD;       // 3*4096*1024
    const size_t ACT = (size_t)GG * BB * TT * DD;  // activation elements

    __bf16* Wqb = (__bf16*)take(WSQ * 2);
    __bf16* Wkb = (__bf16*)take(WSQ * 2);
    __bf16* Wvb = (__bf16*)take(WSQ * 2);
    __bf16* Wob = (__bf16*)take(WSQ * 2);
    __bf16* W1b = (__bf16*)take(WF * 2);
    __bf16* W2b = (__bf16*)take(WF * 2);
    float*  sil = (float*)take((size_t)BB * EE * 4);
    float*  modx  = (float*)take((size_t)GG * BB * 2 * DD * 4);
    float*  modxf = (float*)take((size_t)GG * BB * 2 * DD * 4);
    float*  modf  = (float*)take((size_t)GG * BB * 2 * DD * 4);
    __bf16* xn  = (__bf16*)take(ACT * 2);
    __bf16* xfn = (__bf16*)take(ACT * 2);
    __bf16* Qb  = (__bf16*)take(ACT * 2);
    __bf16* Kb  = (__bf16*)take(ACT * 2);
    __bf16* Vb  = (__bf16*)take(ACT * 2);
    __bf16* Ob  = (__bf16*)take(ACT * 2);
    float*  xmid = (float*)take(ACT * 4);
    __bf16* fn  = (__bf16*)take(ACT * 2);
    __bf16* hdn = (__bf16*)take((size_t)GG * BB * TT * FF * 2);

    // 1) weights -> bf16
    cvt_bf16_kernel<<<(int)(WSQ / 1024), 256, 0, stream>>>(Wq, Wqb, (int)WSQ);
    cvt_bf16_kernel<<<(int)(WSQ / 1024), 256, 0, stream>>>(Wk, Wkb, (int)WSQ);
    cvt_bf16_kernel<<<(int)(WSQ / 1024), 256, 0, stream>>>(Wv, Wvb, (int)WSQ);
    cvt_bf16_kernel<<<(int)(WSQ / 1024), 256, 0, stream>>>(Wo, Wob, (int)WSQ);
    cvt_bf16_kernel<<<(int)(WF  / 1024), 256, 0, stream>>>(W1, W1b, (int)WF);
    cvt_bf16_kernel<<<(int)(WF  / 1024), 256, 0, stream>>>(W2, W2b, (int)WF);

    // 2) modulation
    silu_kernel<<<(BB * EE) / 256, 256, 0, stream>>>(emb, sil, BB * EE);
    int modBlocks = (GG * BB * 2 * DD) / 256;
    mod_kernel<<<modBlocks, 256, 0, stream>>>(sil, Wemb_x,  bemb_x,  modx,  2 * DD);
    mod_kernel<<<modBlocks, 256, 0, stream>>>(sil, Wemb_xf, bemb_xf, modxf, 2 * DD);
    mod_kernel<<<modBlocks, 256, 0, stream>>>(sil, Wemb_f,  bemb_f,  modf,  2 * DD);

    // 3) AdaLN
    adaln_kernel<<<GG * BB * TT, 256, 0, stream>>>(x,  modx,  xn);
    adaln_kernel<<<GG * BB * NN, 256, 0, stream>>>(xf, modxf, xfn);

    // 4) Q/K/V projections (1/sqrt(HD) folded into Q)
    dim3 gProj(DD / 128, GM / 128, GG);
    gemm_bf16_kernel<<<gProj, 256, 0, stream>>>(xn,  Wqb, bq, nullptr, 0.125f, 0, Qb, nullptr, DD, DD);
    gemm_bf16_kernel<<<gProj, 256, 0, stream>>>(xfn, Wkb, bk, nullptr, 1.0f,   0, Kb, nullptr, DD, DD);
    gemm_bf16_kernel<<<gProj, 256, 0, stream>>>(xfn, Wvb, bv, nullptr, 1.0f,   0, Vb, nullptr, DD, DD);

    // 5) joint attention across branches
    attn_kernel<<<dim3(QTOT / 128, HH, BB), 256, 0, stream>>>(Qb, Kb, Vb, Ob);

    // 6) output projection + residual -> xmid (fp32)
    gemm_bf16_kernel<<<gProj, 256, 0, stream>>>(Ob, Wob, bo, x, 1.0f, 0, nullptr, xmid, DD, DD);

    // 7) FFN with AdaLN + residual
    adaln_kernel<<<GG * BB * TT, 256, 0, stream>>>(xmid, modf, fn);
    dim3 gFF1(FF / 128, GM / 128, GG);
    gemm_bf16_kernel<<<gFF1, 256, 0, stream>>>(fn, W1b, b1, nullptr, 1.0f, 1, hdn, nullptr, FF, DD);
    dim3 gFF2(DD / 128, GM / 128, GG);
    gemm_bf16_kernel<<<gFF2, 256, 0, stream>>>(hdn, W2b, b2, xmid, 1.0f, 0, nullptr, (float*)d_out, DD, FF);
}
